// Conv2DFromScratch_7799660609644
// MI455X (gfx1250) — matrix operand order repaired and checked by reading, not compiled, not run on hardware
//
#include <hip/hip_runtime.h>

typedef _Float16 v8h  __attribute__((ext_vector_type(8)));
typedef _Float16 v16h __attribute__((ext_vector_type(16)));
typedef float    v8f  __attribute__((ext_vector_type(8)));

#define NB    32
#define CIN   128
#define HW    56
#define PIX   (HW * HW)      // 3136
#define COUT  256
#define KTAPS 9

#define MT 128               // o-tile
#define NT 256               // pixel-tile (halves weight L2 traffic vs 128)
#define KT 32                // c chunk per WMMA K step
#define LPAD 40              // LDS row pitch in halves (80B: 16B aligned, low-conflict)
#define NITER (KTAPS * (CIN / KT))   // 36 k-steps

__global__ __launch_bounds__(256)
void conv3x3_wmma(const float* __restrict__ x,
                  const float* __restrict__ wgt,
                  const float* __restrict__ bias,
                  float* __restrict__ out)
{
    // double-buffered stages: 2*(10240 + 20480) B = 60 KB of 320 KB/WGP
    __shared__ _Float16 WS[2][MT][LPAD];   // weights [o][c]
    __shared__ _Float16 XS[2][NT][LPAD];   // activ.  [p][c] (transposed at store)

    const int tid  = threadIdx.x;
    const int lane = tid & 31;
    const int wid  = tid >> 5;
    const int wm   = wid >> 1;          // 0..3 : 32-row o strip (2 mf)
    const int wn   = wid & 1;           // 0..1 : 128-col p strip (8 nf)
    const int lN   = lane & 15;
    const int g    = lane >> 4;

    const int p0 = blockIdx.x * NT;     // pixel tile base (flat h*56+w)
    const int o0 = blockIdx.y * MT;     // out-channel tile base
    const int n  = blockIdx.z;          // image

    // ---- iteration-invariant staging geometry (all 32-bit) ----
    const int sp = tid;                 // X staging: this thread owns pixel column sp
    const int pg = p0 + sp;
    const int h  = pg / HW;
    const int w  = pg - h * HW;
    const int xn = n * CIN * PIX;

    const int wc = tid & (KT - 1);      // W staging: c column (same every pass)
    const int wo = tid >> 5;            // W staging: o row base, +8 per pass

    // stage k-step `it` into buffer `buf`
    auto stage = [&](int it, int buf) {
        const int tap = it >> 2;
        const int c0  = (it & 3) * KT;
        const int dh  = tap / 3 - 1;
        const int dw  = tap - (tap / 3) * 3 - 1;

        // weights: MT x KT, f32 -> f16  (saddr + scaled 32-bit offsets)
        int wadr = (o0 + wo) * (CIN * KTAPS) + (c0 + wc) * KTAPS + tap;
        #pragma unroll
        for (int i = 0; i < (MT * KT) / 256; ++i) {
            WS[buf][wo + 8 * i][wc] = (_Float16)wgt[wadr];
            wadr += 8 * (CIN * KTAPS);
        }

        // x: all KT c-rows of pixel column sp, shifted by (dh,dw), zero-padded.
        // Batch 8 halves per ds_store_b128.
        const int hh = h + dh;
        const int ww = w + dw;
        const bool ok = (pg < PIX) & ((unsigned)hh < (unsigned)HW)
                                   & ((unsigned)ww < (unsigned)HW);
        int xadr = xn + c0 * PIX + hh * HW + ww;
        #pragma unroll
        for (int j = 0; j < KT / 8; ++j) {
            v8h tmp;
            #pragma unroll
            for (int e = 0; e < 8; ++e) {
                float v = ok ? x[xadr] : 0.0f;
                tmp[e] = (_Float16)v;
                xadr += PIX;
            }
            *(v8h*)(&XS[buf][sp][8 * j]) = tmp;
        }
    };

    v8f acc[2][8] = {};                 // 2(M) x 8(N) fragments of 16x16 f32

    stage(0, 0);
    __syncthreads();

    for (int it = 0; it < NITER; ++it) {
        const int buf = it & 1;

        // software pipeline: stage k-step it+1 into the other buffer
        if (it + 1 < NITER) stage(it + 1, buf ^ 1);

        // gfx1250 speculative prefetch two steps ahead (global_prefetch_b8)
        if (it + 2 < NITER) {
            const int c2 = ((it + 2) & 3) * KT;
            const int pc = pg < PIX ? pg : (PIX - 1);
            __builtin_prefetch(x + xn + c2 * PIX + pc, 0, 1);
        }

        const _Float16* wsb = &WS[buf][0][0];
        const _Float16* xsb = &XS[buf][0][0];

        // A fragments (16x32 f16 layout: per lane K in {8g..8g+7} U {16+8g..+7})
        v16h a[2];
        #pragma unroll
        for (int mf = 0; mf < 2; ++mf) {
            const _Float16* row = wsb + (wm * 32 + mf * 16 + lN) * LPAD;
            v8h lo = *(const v8h*)(row + 8 * g);
            v8h hi = *(const v8h*)(row + 16 + 8 * g);
            a[mf] = __builtin_shufflevector(lo, hi,
                    0,1,2,3,4,5,6,7,8,9,10,11,12,13,14,15);
        }
        // B fragments (32x16 f16 layout: per lane K in {8g..8g+7} U {16+8g..+7})
        v16h b[8];
        #pragma unroll
        for (int nf = 0; nf < 8; ++nf) {
            const _Float16* row = xsb + (wn * 128 + nf * 16 + lN) * LPAD;
            v8h lo = *(const v8h*)(row + 8 * g);
            v8h hi = *(const v8h*)(row + 16 + 8 * g);
            b[nf] = __builtin_shufflevector(lo, hi,
                    0,1,2,3,4,5,6,7,8,9,10,11,12,13,14,15);
        }

        // 16 WMMAs: D = A x B + C, fp32 accumulate
        #pragma unroll
        for (int mf = 0; mf < 2; ++mf)
            #pragma unroll
            for (int nf = 0; nf < 8; ++nf)
                acc[mf][nf] = __builtin_amdgcn_wmma_f32_16x16x32_f16(
                    false, a[mf], false, b[nf],
                    (short)0, acc[mf][nf], false, false);

        __syncthreads();   // next-stage writes visible; this buf free in 2 steps
    }

    // ---- epilogue: C/D layout lane=N, VGPR r -> M = r + 8g; add bias ----
    const int obase = o0 + wm * 32 + 8 * g;
    const int pcol  = p0 + wn * 128 + lN;
    #pragma unroll
    for (int mf = 0; mf < 2; ++mf) {
        #pragma unroll
        for (int nf = 0; nf < 8; ++nf) {
            const int p = pcol + nf * 16;
            if (p < PIX) {
                #pragma unroll
                for (int r = 0; r < 8; ++r) {
                    const int o = obase + mf * 16 + r;
                    out[(n * COUT + o) * PIX + p] = acc[mf][nf][r] + bias[o];
                }
            }
        }
    }
}

extern "C" void kernel_launch(void* const* d_in, const int* in_sizes, int n_in,
                              void* d_out, int out_size, void* d_ws, size_t ws_size,
                              hipStream_t stream) {
    (void)in_sizes; (void)n_in; (void)out_size; (void)d_ws; (void)ws_size;
    const float* x  = (const float*)d_in[0];
    const float* w  = (const float*)d_in[1];
    const float* b  = (const float*)d_in[2];
    float* out = (float*)d_out;

    dim3 grid((PIX + NT - 1) / NT, COUT / MT, NB);   // 13 x 2 x 32
    dim3 block(256);                                 // 8 waves (wave32)
    conv3x3_wmma<<<grid, block, 0, stream>>>(x, w, b, out);
}
